// SelfAttention_6906307412425
// MI455X (gfx1250) — compile-verified, matrix operand order checked by reading, not run on hardware
//
#include <hip/hip_runtime.h>

// Problem constants (from reference): B=4, S=2048, E=768, H=12, hd=64
#define BB 4
#define SS 2048
#define EE 768
#define HH 12
#define HD 64
#define NROW (BB*SS)   // 8192 token rows
#define F3  (3*EE)     // 2304 fused qkv output features

typedef __attribute__((ext_vector_type(16))) __bf16 v16bf;
typedef __attribute__((ext_vector_type(8)))  __bf16 v8bf;
typedef __attribute__((ext_vector_type(8)))  float  v8f;

__device__ __forceinline__ v8f wmma_bf16(v16bf a, v16bf b, v8f c) {
    // D = A(16x32 bf16) x B(32x16 bf16) + C(16x16 f32)
    return __builtin_amdgcn_wmma_f32_16x16x32_bf16(
        /*neg_a=*/false, a, /*neg_b=*/false, b,
        /*c_mod=*/(short)0, c, /*reuse_a=*/false, /*reuse_b=*/false);
}

// A/B fragment from bf16 memory (global or LDS).
// Per ISA 7.12.2: lanes 0-15 hold K {k0+0..7, k0+16..23}; lanes 16-31 hold {k0+8..15, k0+24..31}.
__device__ __forceinline__ v16bf frag_from_bf16(const __bf16* row, int k0, int half) {
    v8bf lo = *(const v8bf*)(row + k0 + 8 * half);
    v8bf hi = *(const v8bf*)(row + k0 + 16 + 8 * half);
    return __builtin_shufflevector(lo, hi, 0,1,2,3,4,5,6,7,8,9,10,11,12,13,14,15);
}

// One k-step worth of GEMM fragments: 2 A rows x 4 B columns (32x64 tile).
struct FragSet { v16bf a0, a1, b0, b1, b2, b3; };

__device__ __forceinline__ void load_set(FragSet& s,
    const __bf16* xr0, const __bf16* xr1, const __bf16* wr, int k0, int half) {
    s.a0 = frag_from_bf16(xr0, k0, half);
    s.a1 = frag_from_bf16(xr1, k0, half);
    s.b0 = frag_from_bf16(wr, k0, half);
    s.b1 = frag_from_bf16(wr + (size_t)16 * EE, k0, half);
    s.b2 = frag_from_bf16(wr + (size_t)32 * EE, k0, half);
    s.b3 = frag_from_bf16(wr + (size_t)48 * EE, k0, half);
}

__device__ __forceinline__ void do_wmma(v8f acc[2][4], const FragSet& s) {
    acc[0][0] = wmma_bf16(s.a0, s.b0, acc[0][0]);
    acc[1][0] = wmma_bf16(s.a1, s.b0, acc[1][0]);
    acc[0][1] = wmma_bf16(s.a0, s.b1, acc[0][1]);
    acc[1][1] = wmma_bf16(s.a1, s.b1, acc[1][1]);
    acc[0][2] = wmma_bf16(s.a0, s.b2, acc[0][2]);
    acc[1][2] = wmma_bf16(s.a1, s.b2, acc[1][2]);
    acc[0][3] = wmma_bf16(s.a0, s.b3, acc[0][3]);
    acc[1][3] = wmma_bf16(s.a1, s.b3, acc[1][3]);
}

// 64B/lane async global->LDS copy (CDNA5, tracked by ASYNCcnt).
// The 24-bit inst offset is added to BOTH the LDS and global addresses.
__device__ __forceinline__ void async_copy64(unsigned lds, unsigned long long ga) {
    asm volatile("global_load_async_to_lds_b128 %0, %1, off"
                 :: "v"(lds), "v"(ga) : "memory");
    asm volatile("global_load_async_to_lds_b128 %0, %1, off offset:16"
                 :: "v"(lds), "v"(ga) : "memory");
    asm volatile("global_load_async_to_lds_b128 %0, %1, off offset:32"
                 :: "v"(lds), "v"(ga) : "memory");
    asm volatile("global_load_async_to_lds_b128 %0, %1, off offset:48"
                 :: "v"(lds), "v"(ga) : "memory");
}

// ---------------------------------------------------------------------------
// Kernel 0: fp32 -> bf16 bulk convert (one pass; removes cvt from GEMM loops)
// ---------------------------------------------------------------------------
__global__ __launch_bounds__(256) void cvt_kernel(
    const float* __restrict__ src, __bf16* __restrict__ dst, int n8) {
    int i = blockIdx.x * 256 + threadIdx.x;
    if (i >= n8) return;
    const float4* p = (const float4*)src + (size_t)i * 2;
    float4 a = p[0], b = p[1];
    v8bf r;
    r[0]=(__bf16)a.x; r[1]=(__bf16)a.y; r[2]=(__bf16)a.z; r[3]=(__bf16)a.w;
    r[4]=(__bf16)b.x; r[5]=(__bf16)b.y; r[6]=(__bf16)b.z; r[7]=(__bf16)b.w;
    *((v8bf*)dst + i) = r;
}

// ---------------------------------------------------------------------------
// Kernel 1: fused QKV projection  qkv[m,f] = sum_e xb[m,e] * wqkvb[f,e]
// 32x64 output tile per wave. k-loop is software-pipelined and 2x unrolled
// with ping-pong fragment sets A/B (no rotation copies): loadB(k+32),
// wmma(A), loadA(k+64), wmma(B).
// Scatter into per-head bf16 layouts: Q[B,H,S,64], K[B,H,S,64], Vt[B,H,64,S].
// ---------------------------------------------------------------------------
__global__ __launch_bounds__(128) void qkv_kernel(
    const __bf16* __restrict__ xb, const __bf16* __restrict__ wqkvb,
    __bf16* __restrict__ Qw, __bf16* __restrict__ Kw, __bf16* __restrict__ Vw) {
    // readfirstlane pins the wave id (and all tile coords) into SGPRs:
    // epilogue region branch becomes a scalar branch, bases use SALU.
    const int wid    = __builtin_amdgcn_readfirstlane(threadIdx.x >> 5);
    const int wave   = blockIdx.x * 4 + wid;
    const int lane   = threadIdx.x & 31;
    const int lane16 = lane & 15;
    const int half   = lane >> 4;
    const int NT = F3 / 64;                 // 36 n-tiles
    const int mt = wave / NT;
    const int nt = wave - mt * NT;
    const int m0 = mt * 32, n0 = nt * 64;

    const __bf16* xrow0 = xb + (size_t)(m0 + lane16) * EE;
    const __bf16* xrow1 = xrow0 + (size_t)16 * EE;
    const __bf16* wrow0 = wqkvb + (size_t)(n0 + lane16) * EE;

    v8f acc[2][4] = {};
    FragSet A, B;
    load_set(A, xrow0, xrow1, wrow0, 0, half);
    int k0 = 0;
    for (; k0 + 64 < EE; k0 += 64) {        // 11 double-steps
        load_set(B, xrow0, xrow1, wrow0, k0 + 32, half);
        do_wmma(acc, A);
        load_set(A, xrow0, xrow1, wrow0, k0 + 64, half);
        do_wmma(acc, B);
    }
    load_set(B, xrow0, xrow1, wrow0, k0 + 32, half);  // last k-step (EE-32)
    do_wmma(acc, A);                                  // k-step EE-64
    do_wmma(acc, B);

    // Scatter. Each 16-wide j-subtile lies entirely inside one Q/K/V region
    // (subtile base and region boundaries are multiples of 16), so the region
    // choice is scalar (fbase/hh/jb are SGPR values).
#pragma unroll
    for (int j = 0; j < 4; ++j) {
        const int fbase = n0 + j * 16;       // scalar
        const int hh = fbase / 192;          // head index (3*hd = 192 per head)
        const int jb = fbase - hh * 192;     // multiple of 16 in [0,192)
        if (jb < 128) {                      // Q or K region, row-major [.,S,64]
            __bf16* base = (jb < 64) ? Qw : Kw;
            const int col = (jb & 63) + lane16;
#pragma unroll
            for (int mm = 0; mm < 2; ++mm)
#pragma unroll
                for (int g = 0; g < 8; ++g) {
                    const int m = m0 + mm * 16 + g + 8 * half;
                    const int b = m >> 11;
                    const int s = m & (SS - 1);
                    base[(((size_t)(b * HH + hh)) * SS + s) * HD + col] =
                        (__bf16)acc[mm][j][g];
                }
        } else {                             // V region, transposed [.,64,S]
            const int dd = (jb - 128) + lane16;
#pragma unroll
            for (int mm = 0; mm < 2; ++mm)
#pragma unroll
                for (int g = 0; g < 8; ++g) {
                    const int m = m0 + mm * 16 + g + 8 * half;
                    const int b = m >> 11;
                    const int s = m & (SS - 1);
                    Vw[(((size_t)(b * HH + hh)) * HD + dd) * SS + s] =
                        (__bf16)acc[mm][j][g];
                }
        }
    }
}

// ---------------------------------------------------------------------------
// Kernel 2: flash attention. Block = 4 waves, one (b,h,qtile-of-64) per block.
// Each wave: 16 q rows x hd=64. K/Vt tiles (64 keys) double-buffered in LDS
// via CDNA5 async global->LDS copies; ASYNCcnt retires in-order so
// s_wait_asynccnt 8 == "previous tile complete" while the next is in flight.
// ---------------------------------------------------------------------------
__global__ __launch_bounds__(128) void attn_kernel(
    const __bf16* __restrict__ Qw, const __bf16* __restrict__ Kw,
    const __bf16* __restrict__ Vw, const int* __restrict__ mask,
    __bf16* __restrict__ attn) {
    __shared__ __bf16 Kls[2][64 * 72];      // K tile: [key][d], padded stride
    __shared__ __bf16 Vls[2][64 * 72];      // Vt tile: [d][key]
    __shared__ __bf16 Pls[4 * 16 * 72];     // per-wave P transpose staging

    const int tid    = threadIdx.x;
    const int wave   = __builtin_amdgcn_readfirstlane(tid >> 5);
    const int lane   = tid & 31;
    const int lane16 = lane & 15;
    const int half   = lane >> 4;
    const int qt = blockIdx.x & 31;         // 32 q-tiles of 64
    const int bh = blockIdx.x >> 5;         // b*H + h
    const int b  = bh / HH;
    const int h  = bh - b * HH;
    const int q0 = qt * 64 + wave * 16;

    const __bf16* Qbase = Qw + (size_t)bh * SS * HD;
    const __bf16* Kbase = Kw + (size_t)bh * SS * HD;
    const __bf16* Vbase = Vw + (size_t)bh * HD * SS;
    const int* mrow = mask + b * SS;

    const __bf16* qrow = Qbase + (size_t)(q0 + lane16) * HD;
    const v16bf aQ0 = frag_from_bf16(qrow, 0, half);
    const v16bf aQ1 = frag_from_bf16(qrow, 32, half);

    v8f acc[4] = {};
    float rm[8], rl[8];
#pragma unroll
    for (int g = 0; g < 8; ++g) { rm[g] = -1e30f; rl[g] = 0.f; }

    const float scl = 0.125f * 1.44269504f; // 1/sqrt(64) * log2(e)
    const int row2 = tid >> 1;              // 0..63 (tile row for coop load)
    const int c0   = (tid & 1) * 32;        // 32 bf16 per thread per row

    // per-thread LDS destinations (low 32 bits of generic ptr == LDS offset)
    unsigned klsA[2], vlsA[2];
#pragma unroll
    for (int p = 0; p < 2; ++p) {
        klsA[p] = (unsigned)(size_t)(&Kls[p][row2 * 72 + c0]);
        vlsA[p] = (unsigned)(size_t)(&Vls[p][row2 * 72 + c0]);
    }
    const __bf16* gK = Kbase + (size_t)row2 * HD + c0;   // advance by 64 rows/iter
    const __bf16* gV = Vbase + (size_t)row2 * SS + c0;   // advance by 64 cols/iter

    // prologue: tile 0 -> buffer 0 (8 async b128 per thread pair of tiles)
    async_copy64(klsA[0], (unsigned long long)(size_t)gK);
    async_copy64(vlsA[0], (unsigned long long)(size_t)gV);

    for (int kt = 0; kt < SS; kt += 64) {
        const int p = (kt >> 6) & 1;
        if (kt + 64 < SS) {                 // prefetch next tile into other buf
            async_copy64(klsA[p ^ 1],
                (unsigned long long)(size_t)(gK + (size_t)(kt + 64) * HD));
            async_copy64(vlsA[p ^ 1],
                (unsigned long long)(size_t)(gV + (kt + 64)));
            asm volatile("s_wait_asynccnt 0x8" ::: "memory"); // cur tile done
        } else {
            asm volatile("s_wait_asynccnt 0x0" ::: "memory");
        }
        __syncthreads();                    // cur tile visible to all waves

        const __bf16* Kbuf = &Kls[p][0];
        const __bf16* Vbuf = &Vls[p][0];

        // logits: S = Q(16x64) @ K^T(64x64) for 4 key sub-tiles of 16
        float t[4][8];
#pragma unroll
        for (int n = 0; n < 4; ++n) {
            const __bf16* krow = Kbuf + (n * 16 + lane16) * 72;
            v8f c = {};
            c = wmma_bf16(aQ0, frag_from_bf16(krow, 0, half), c);
            c = wmma_bf16(aQ1, frag_from_bf16(krow, 32, half), c);
            const int key = kt + n * 16 + lane16;   // this lane's key column
            const bool keep = (mrow[key] != 0);
#pragma unroll
            for (int g = 0; g < 8; ++g)
                t[n][g] = keep ? c[g] * scl : -100000.f;
        }

        // online softmax per output row g (row spread across 16 lanes of a half)
#pragma unroll
        for (int g = 0; g < 8; ++g) {
            float mt = fmaxf(fmaxf(t[0][g], t[1][g]), fmaxf(t[2][g], t[3][g]));
            mt = fmaxf(mt, __shfl_xor(mt, 1, 32));
            mt = fmaxf(mt, __shfl_xor(mt, 2, 32));
            mt = fmaxf(mt, __shfl_xor(mt, 4, 32));
            mt = fmaxf(mt, __shfl_xor(mt, 8, 32));
            const float nm    = fmaxf(rm[g], mt);
            const float alpha = exp2f(rm[g] - nm);
            float ps = 0.f;
#pragma unroll
            for (int n = 0; n < 4; ++n) {
                const float pp = exp2f(t[n][g] - nm);
                t[n][g] = pp; ps += pp;
            }
            ps += __shfl_xor(ps, 1, 32);
            ps += __shfl_xor(ps, 2, 32);
            ps += __shfl_xor(ps, 4, 32);
            ps += __shfl_xor(ps, 8, 32);
            rl[g] = rl[g] * alpha + ps;
            rm[g] = nm;
#pragma unroll
            for (int d = 0; d < 4; ++d) acc[d][g] *= alpha;
        }

        // transpose P (C-frag layout -> A-frag layout) via per-wave LDS staging
        __bf16* Pw = &Pls[wave * 16 * 72];
#pragma unroll
        for (int n = 0; n < 4; ++n)
#pragma unroll
            for (int g = 0; g < 8; ++g)
                Pw[(g + 8 * half) * 72 + n * 16 + lane16] = (__bf16)t[n][g];
        asm volatile("s_wait_dscnt 0x0" ::: "memory");  // same-wave LDS RAW
        const __bf16* prow = Pw + lane16 * 72;
        const v16bf aP0 = frag_from_bf16(prow, 0, half);
        const v16bf aP1 = frag_from_bf16(prow, 32, half);

        // O += P(16x64) @ V(64x64), B-frags read Vt rows contiguously
#pragma unroll
        for (int d = 0; d < 4; ++d) {
            const __bf16* vrow = Vbuf + (d * 16 + lane16) * 72;
            acc[d] = wmma_bf16(aP0, frag_from_bf16(vrow, 0, half), acc[d]);
            acc[d] = wmma_bf16(aP1, frag_from_bf16(vrow, 32, half), acc[d]);
        }
        __syncthreads();                    // reads done before buf is reused
    }

    // normalize and store per-head output back into [B,S,E] bf16
#pragma unroll
    for (int d = 0; d < 4; ++d)
#pragma unroll
        for (int g = 0; g < 8; ++g) {
            const int q   = q0 + g + 8 * half;
            const int col = h * HD + d * 16 + lane16;
            attn[((size_t)b * SS + q) * EE + col] = (__bf16)(acc[d][g] / rl[g]);
        }
}

// ---------------------------------------------------------------------------
// Kernel 3: output projection  out[m,f] = sum_e attn[m,e] * woutb[f,e]
// 32x64 tile per wave, ping-pong pipelined k-loop, fp32 output.
// ---------------------------------------------------------------------------
__global__ __launch_bounds__(128) void proj_kernel(
    const __bf16* __restrict__ attn, const __bf16* __restrict__ woutb,
    float* __restrict__ out) {
    const int wid    = __builtin_amdgcn_readfirstlane(threadIdx.x >> 5);
    const int wave   = blockIdx.x * 4 + wid;
    const int lane   = threadIdx.x & 31;
    const int lane16 = lane & 15;
    const int half   = lane >> 4;
    const int NT = EE / 64;                 // 12 n-tiles
    const int mt = wave / NT;
    const int nt = wave - mt * NT;
    const int m0 = mt * 32, n0 = nt * 64;

    const __bf16* arow0 = attn + (size_t)(m0 + lane16) * EE;
    const __bf16* arow1 = arow0 + (size_t)16 * EE;
    const __bf16* wrow0 = woutb + (size_t)(n0 + lane16) * EE;

    v8f acc[2][4] = {};
    FragSet A, B;
    load_set(A, arow0, arow1, wrow0, 0, half);
    int k0 = 0;
    for (; k0 + 64 < EE; k0 += 64) {
        load_set(B, arow0, arow1, wrow0, k0 + 32, half);
        do_wmma(acc, A);
        load_set(A, arow0, arow1, wrow0, k0 + 64, half);
        do_wmma(acc, B);
    }
    load_set(B, arow0, arow1, wrow0, k0 + 32, half);
    do_wmma(acc, A);
    do_wmma(acc, B);

#pragma unroll
    for (int j = 0; j < 4; ++j)
#pragma unroll
        for (int mm = 0; mm < 2; ++mm)
#pragma unroll
            for (int g = 0; g < 8; ++g) {
                const int m = m0 + mm * 16 + g + 8 * half;
                const int f = n0 + j * 16 + lane16;
                out[(size_t)m * EE + f] = acc[mm][j][g];
            }
}

// ---------------------------------------------------------------------------
extern "C" void kernel_launch(void* const* d_in, const int* in_sizes, int n_in,
                              void* d_out, int out_size, void* d_ws, size_t ws_size,
                              hipStream_t stream) {
    const float* x    = (const float*)d_in[0];
    const float* wqkv = (const float*)d_in[1];
    const float* wout = (const float*)d_in[2];
    const int*   mask = (const int*)d_in[3];

    char* ws = (char*)d_ws;
    size_t off = 0;
    __bf16* xb    = (__bf16*)(ws + off); off += (size_t)NROW * EE * 2; // 12.6 MB
    __bf16* wqkvb = (__bf16*)(ws + off); off += (size_t)F3   * EE * 2; //  3.5 MB
    __bf16* woutb = (__bf16*)(ws + off); off += (size_t)EE   * EE * 2; //  1.2 MB
    const size_t tbytes = (size_t)BB * HH * SS * HD * 2;               // 12.6 MB
    __bf16* Qw   = (__bf16*)(ws + off); off += tbytes;
    __bf16* Kw   = (__bf16*)(ws + off); off += tbytes;
    __bf16* Vw   = (__bf16*)(ws + off); off += tbytes;
    __bf16* attn = (__bf16*)(ws + off); off += tbytes;                 // ~68 MB total

    // fp32 -> bf16 conversions (memory-bound single pass)
    cvt_kernel<<<(NROW * EE / 8) / 256, 256, 0, stream>>>(x, xb, NROW * EE / 8);
    cvt_kernel<<<(F3 * EE / 8) / 256, 256, 0, stream>>>(wqkv, wqkvb, F3 * EE / 8);
    cvt_kernel<<<(EE * EE / 8) / 256, 256, 0, stream>>>(wout, woutb, EE * EE / 8);

    {   // QKV projection: (8192/32)*(2304/64) = 9216 wave-tiles, 4 waves/block
        const int blocks = (NROW / 32) * (F3 / 64) / 4;
        qkv_kernel<<<blocks, 128, 0, stream>>>(xb, wqkvb, Qw, Kw, Vw);
    }
    {   // flash attention: B*H*(S/64) = 1536 blocks
        attn_kernel<<<BB * HH * (SS / 64), 128, 0, stream>>>(Qw, Kw, Vw, mask, attn);
    }
    {   // output projection: (8192/32)*(768/64) = 3072 wave-tiles, 4 waves/block
        const int blocks = (NROW / 32) * (EE / 64) / 4;
        proj_kernel<<<blocks, 128, 0, stream>>>(attn, woutb, (float*)d_out);
    }
}
